// BEEncdecMultiheadAttn_4544075399699
// MI455X (gfx1250) — compile-verified, matrix operand order checked by reading, not run on hardware
//
#include <hip/hip_runtime.h>

typedef __bf16 v16bf __attribute__((ext_vector_type(16)));
typedef __bf16 v8bf  __attribute__((ext_vector_type(8)));
typedef float  v8f   __attribute__((ext_vector_type(8)));

#define Tq 1024
#define Tk 1024
#define Bsz 8
#define Hdim 1024
#define NHEADS 16
#define HD 64
#define Mrows 8192            // Tq*B == Tk*B
#define NTOT (Bsz*NHEADS)     // 128 batched heads

__device__ __forceinline__ v8f vzero8() {
    v8f z;
#pragma unroll
    for (int i = 0; i < 8; ++i) z[i] = 0.0f;
    return z;
}

// Load a 16x32 (or 32x16) bf16 WMMA operand slice for this lane:
// two contiguous 16B chunks -> v16bf (8 VGPRs). Works for global or LDS ptrs.
__device__ __forceinline__ v16bf ld_op(const __bf16* p0, const __bf16* p1) {
    v8bf lo = *(const v8bf*)p0;
    v8bf hi = *(const v8bf*)p1;
    return __builtin_shufflevector(lo, hi, 0, 1, 2, 3, 4, 5, 6, 7,
                                   8, 9, 10, 11, 12, 13, 14, 15);
}

__device__ __forceinline__ v8f wmma_bf16(v16bf a, v16bf b, v8f c) {
    return __builtin_amdgcn_wmma_f32_16x16x32_bf16(false, a, false, b,
                                                   (short)0, c, false, false);
}

// CDNA5 async global->LDS copy, 16B per lane, tracked by ASYNCcnt.
// Generic LDS address low 32 bits == LDS byte offset (aperture mapping).
__device__ __forceinline__ void async_copy_b128(const __bf16* g, __bf16* l) {
    unsigned int loff = (unsigned int)(uintptr_t)l;
    unsigned long long ga = (unsigned long long)(uintptr_t)g;
    asm volatile("global_load_async_to_lds_b128 %0, %1, off"
                 :: "v"(loff), "v"(ga) : "memory");
}
__device__ __forceinline__ void wait_async0() {
    asm volatile("s_wait_asynccnt 0" ::: "memory");
}

// ---------------------------------------------------------------------------
// Stage 0a: x_bf[m,h] = bf16( x[m,h] * r[b,h] ),  m = t*B+b, H=1024
// ---------------------------------------------------------------------------
__global__ __launch_bounds__(256) void cast_scale_x(const float* __restrict__ x,
                                                    const float* __restrict__ r,
                                                    __bf16* __restrict__ o) {
    size_t i = (size_t)blockIdx.x * 256 + threadIdx.x;   // 8M elements
    int h = (int)(i & (Hdim - 1));
    int b = (int)((i >> 10) & (Bsz - 1));
    o[i] = (__bf16)(x[i] * r[b * Hdim + h]);
}

// Stage 0b: plain f32 -> bf16 weight cast
__global__ __launch_bounds__(256) void cast_w(const float* __restrict__ w,
                                              __bf16* __restrict__ o, int n) {
    int i = blockIdx.x * 256 + threadIdx.x;
    if (i < n) o[i] = (__bf16)w[i];
}

// ---------------------------------------------------------------------------
// WMMA GEMM:  C[M,N] = A[M,K] * W[N,K]^T   (A,W bf16, K contiguous for both)
// Block: 256 thr = 8 waves stacked in M. Block tile 128(M) x 64(N).
// Wave tile: 16(M) x 64(N) = 4 accumulators, K step 32.
// W 64x32 tile is shared by all 8 waves -> staged in LDS via async copies,
// double-buffered (prefetch chunk c+1 while computing chunk c).
// MODE 0: Q proj   -> q_bf[n][q][d]   (apply s_q, b_q, fold hd^-0.5)
// MODE 1: KV proj  -> k_bf[n][k][d], vT[n][d][k]  (apply s_kv, b_kv)
// MODE 2: out proj -> f32 out[m][o]  (+ b_o)
// ---------------------------------------------------------------------------
template <int MODE>
__global__ __launch_bounds__(256) void gemm_wmma(const __bf16* __restrict__ A,
                                                 const __bf16* __restrict__ W,
                                                 const float* __restrict__ svec,
                                                 const float* __restrict__ bvec,
                                                 void* __restrict__ out,
                                                 void* __restrict__ out2,
                                                 int N, int K) {
    __shared__ __bf16 wsm[2][64 * 32];   // [buf][n(64) x k(32)] = 4KB each

    const int tid  = threadIdx.x;
    const int lane = tid & 31;
    const int wave = tid >> 5;
    const int hh   = lane >> 4;    // half-wave
    const int l16  = lane & 15;
    const int mbase = blockIdx.x * 128 + wave * 16;
    const int nbase = blockIdx.y * 64;

    // per-thread slice of the 4KB W tile: 256 threads x 16B
    const int wrow = tid >> 2;         // 0..63  (n within tile)
    const int wcol = (tid & 3) * 8;    // 0,8,16,24 (k element offset)
    const __bf16* wsrc = W + (size_t)(nbase + wrow) * K + wcol;

    v8f acc[4];
#pragma unroll
    for (int j = 0; j < 4; ++j) acc[j] = vzero8();

    const __bf16* arow = A + (size_t)(mbase + l16) * K;
    const int NC = K >> 5;

    // prologue: prefetch chunk 0 into buffer 0
    async_copy_b128(wsrc, &wsm[0][wrow * 32 + wcol]);

    int buf = 0;
    for (int c = 0; c < NC; ++c) {
        wait_async0();          // this thread's LDS writes landed
        __syncthreads();        // everyone's landed; prev buffer fully read

        if (c + 1 < NC)
            async_copy_b128(wsrc + (c + 1) * 32,
                            &wsm[buf ^ 1][wrow * 32 + wcol]);

        const int kc = c * 32;
        v16bf av = ld_op(arow + kc + 8 * hh, arow + kc + 16 + 8 * hh);
        const __bf16* wt = wsm[buf];
#pragma unroll
        for (int j = 0; j < 4; ++j) {
            const __bf16* wr = wt + (j * 16 + l16) * 32;
            v16bf bv = ld_op(wr + 8 * hh, wr + 16 + 8 * hh);
            acc[j] = wmma_bf16(av, bv, acc[j]);
        }
        buf ^= 1;
    }

    // D layout: per lane, column o = nbase + j*16 + l16; rows m = mbase + v + 8*hh
    if (MODE == 0) {
        __bf16* qb = (__bf16*)out;
#pragma unroll
        for (int j = 0; j < 4; ++j) {
            const int o = nbase + j * 16 + l16;
            const int head = o >> 6, d = o & 63;
            const float bias = bvec[o];
#pragma unroll
            for (int v = 0; v < 8; ++v) {
                const int m = mbase + v + 8 * hh;
                const int t = m >> 3, b = m & 7;
                float val = (acc[j][v] * svec[b * Hdim + o] + bias) * 0.125f;  // * hd^-0.5
                qb[((size_t)((b * NHEADS + head) * Tq + t)) * HD + d] = (__bf16)val;
            }
        }
    } else if (MODE == 1) {
        __bf16* kb = (__bf16*)out;
        __bf16* vt = (__bf16*)out2;
#pragma unroll
        for (int j = 0; j < 4; ++j) {
            const int o = nbase + j * 16 + l16;
            const int head = o >> 7, which = (o >> 6) & 1, d = o & 63;
            const float bias = bvec[o];
#pragma unroll
            for (int v = 0; v < 8; ++v) {
                const int m = mbase + v + 8 * hh;
                const int t = m >> 3, b = m & 7;
                float val = acc[j][v] * svec[b * (2 * Hdim) + o] + bias;
                const int n = b * NHEADS + head;
                if (which == 0)
                    kb[((size_t)(n * Tk + t)) * HD + d] = (__bf16)val;
                else
                    vt[((size_t)(n * HD + d)) * Tk + t] = (__bf16)val;
            }
        }
    } else {
        float* of = (float*)out;
#pragma unroll
        for (int j = 0; j < 4; ++j) {
            const int o = nbase + j * 16 + l16;
            const float bias = bvec[o];
#pragma unroll
            for (int v = 0; v < 8; ++v) {
                const int m = mbase + v + 8 * hh;
                of[(size_t)m * Hdim + o] = acc[j][v] + bias;
            }
        }
    }
}

// ---------------------------------------------------------------------------
// Flash attention. Block = 8 waves, all on the same head n; wave w owns q-tile
// qt = blockIdx.y*8 + w. K/V 32-key chunks are shared by all 8 waves -> staged
// in LDS (async, double-buffered). Scores computed transposed:
// S^T(16k x 16q) = Ktile(16k x 64d) x Qtile^T; the S^T D-layout is exactly the
// B-operand layout for ctx^T += V^T(16d x 32k) x P^T(32k x 16q), so exp'd
// probabilities become the bf16 B operand lane-locally (no shuffles).
// ---------------------------------------------------------------------------
__global__ __launch_bounds__(256) void attn_wmma(const __bf16* __restrict__ qb,
                                                 const __bf16* __restrict__ kb,
                                                 const __bf16* __restrict__ vt,
                                                 __bf16* __restrict__ ctx) {
    __shared__ __bf16 ksm[2][32 * 64];   // [buf][k(32) x d(64)] = 4KB each
    __shared__ __bf16 vsm[2][64 * 32];   // [buf][d(64) x k(32)] = 4KB each

    const int tid  = threadIdx.x;
    const int lane = tid & 31;
    const int wave = tid >> 5;
    const int hh   = lane >> 4;
    const int l16  = lane & 15;
    const int n    = blockIdx.x;                 // 0..127 batched head
    const int qt   = blockIdx.y * 8 + wave;      // 0..63 q tile
    const int q    = qt * 16 + l16;

    // per-thread slices of the K and V^T tiles (256 threads x 16B each)
    const int krow = tid >> 3, kcol = (tid & 7) * 8;      // K: 32 x 128B rows
    const int vrow = tid >> 2, vcol = (tid & 3) * 8;      // V^T: 64 x 64B rows
    const __bf16* ksrc = kb + (size_t)(n * Tk + krow) * HD + kcol;
    const __bf16* vsrc = vt + (size_t)(n * HD + vrow) * Tk + vcol;

    // Loop-invariant Q operands (B matrix, N=q, K-dim = d in chunks of 32)
    const __bf16* qrow = qb + ((size_t)(n * Tq + q)) * HD;
    const v16bf bq0 = ld_op(qrow + 0 + 8 * hh, qrow + 16 + 8 * hh);
    const v16bf bq1 = ld_op(qrow + 32 + 8 * hh, qrow + 48 + 8 * hh);

    v8f ctxa[4];
#pragma unroll
    for (int j = 0; j < 4; ++j) ctxa[j] = vzero8();
    float m_i = -1e30f, l_i = 0.0f;

    // prologue: prefetch chunk 0
    async_copy_b128(ksrc, &ksm[0][krow * 64 + kcol]);
    async_copy_b128(vsrc, &vsm[0][vrow * 32 + vcol]);

    int buf = 0;
    for (int c = 0; c < (Tk >> 5); ++c) {
        wait_async0();
        __syncthreads();

        if (c + 1 < (Tk >> 5)) {
            const int kn = (c + 1) * 32;
            async_copy_b128(ksrc + (size_t)kn * HD, &ksm[buf ^ 1][krow * 64 + kcol]);
            async_copy_b128(vsrc + kn, &vsm[buf ^ 1][vrow * 32 + vcol]);
        }

        // K tiles as A operand (M=k rows, K-dim=d), from LDS
        const __bf16* ks = ksm[buf];
        const __bf16* kr0 = ks + l16 * 64;
        const __bf16* kr1 = ks + (16 + l16) * 64;
        v16bf ak00 = ld_op(kr0 + 8 * hh, kr0 + 16 + 8 * hh);
        v16bf ak01 = ld_op(kr0 + 32 + 8 * hh, kr0 + 48 + 8 * hh);
        v16bf ak10 = ld_op(kr1 + 8 * hh, kr1 + 16 + 8 * hh);
        v16bf ak11 = ld_op(kr1 + 32 + 8 * hh, kr1 + 48 + 8 * hh);

        v8f s0 = vzero8();
        s0 = wmma_bf16(ak00, bq0, s0);
        s0 = wmma_bf16(ak01, bq1, s0);
        v8f s1 = vzero8();
        s1 = wmma_bf16(ak10, bq0, s1);
        s1 = wmma_bf16(ak11, bq1, s1);

        // Online softmax over column q (per lane l16); k rows are lane-local
        // VGPRs + the mirror half-wave.
        float mc = -1e30f;
#pragma unroll
        for (int v = 0; v < 8; ++v) {
            mc = fmaxf(mc, s0[v]);
            mc = fmaxf(mc, s1[v]);
        }
        mc = fmaxf(mc, __shfl_xor(mc, 16, 32));
        const float m_new = fmaxf(m_i, mc);
        const float alpha = __expf(m_i - m_new);

        float p0[8], p1[8], ls = 0.0f;
#pragma unroll
        for (int v = 0; v < 8; ++v) {
            p0[v] = __expf(s0[v] - m_new);
            p1[v] = __expf(s1[v] - m_new);
            ls += p0[v] + p1[v];
        }
        ls += __shfl_xor(ls, 16, 32);
        l_i = l_i * alpha + ls;
        m_i = m_new;

#pragma unroll
        for (int j = 0; j < 4; ++j)
#pragma unroll
            for (int v = 0; v < 8; ++v) ctxa[j][v] *= alpha;

        // Pack P^T as bf16 B operand: element e<8 -> k=kc+8*hh+e (s0 vgpr e),
        // e>=8 -> k=kc+16+8*hh+(e-8) (s1 vgpr e-8). Purely lane-local.
        v16bf pb;
#pragma unroll
        for (int e = 0; e < 8; ++e) pb[e] = (__bf16)p0[e];
#pragma unroll
        for (int e = 8; e < 16; ++e) pb[e] = (__bf16)p1[e - 8];

        // ctx^T += V^T x P^T   (A: M=d rows from LDS V^T tile)
        const __bf16* vs = vsm[buf];
#pragma unroll
        for (int dt = 0; dt < 4; ++dt) {
            const __bf16* vr = vs + (dt * 16 + l16) * 32;
            v16bf avv = ld_op(vr + 8 * hh, vr + 16 + 8 * hh);
            ctxa[dt] = wmma_bf16(avv, pb, ctxa[dt]);
        }
        buf ^= 1;
    }

    const float inv = 1.0f / l_i;
    const int b = n >> 4, head = n & 15;
    // ctx^T D-layout: col q per lane; rows d = dt*16 + v + 8*hh -> 8 contiguous
    // bf16 per accumulator => one 16B store each into ctx[t=q][b][head*64+d].
#pragma unroll
    for (int dt = 0; dt < 4; ++dt) {
        v8bf o8;
#pragma unroll
        for (int v = 0; v < 8; ++v) o8[v] = (__bf16)(ctxa[dt][v] * inv);
        __bf16* dst = ctx + ((size_t)(q * Bsz + b)) * Hdim + head * HD + dt * 16 + 8 * hh;
        *(v8bf*)dst = o8;
    }
}

// ---------------------------------------------------------------------------
// Host launcher
// ---------------------------------------------------------------------------
extern "C" void kernel_launch(void* const* d_in, const int* in_sizes, int n_in,
                              void* d_out, int out_size, void* d_ws, size_t ws_size,
                              hipStream_t stream) {
    const float* inputs_q  = (const float*)d_in[0];
    const float* inputs_kv = (const float*)d_in[1];
    const float* w_q  = (const float*)d_in[2];
    const float* b_q  = (const float*)d_in[3];
    const float* w_kv = (const float*)d_in[4];
    const float* b_kv = (const float*)d_in[5];
    const float* w_o  = (const float*)d_in[6];
    const float* b_o  = (const float*)d_in[7];
    const float* r_q  = (const float*)d_in[8];
    const float* s_q  = (const float*)d_in[9];
    const float* r_kv = (const float*)d_in[10];
    const float* s_kv = (const float*)d_in[11];

    char* ws = (char*)d_ws;
    const size_t MB = 1024ull * 1024ull;
    __bf16* wq_bf  = (__bf16*)(ws + 0 * MB);    // 2 MB
    __bf16* wkv_bf = (__bf16*)(ws + 2 * MB);    // 4 MB
    __bf16* wo_bf  = (__bf16*)(ws + 6 * MB);    // 2 MB
    __bf16* xq_bf  = (__bf16*)(ws + 8 * MB);    // 16 MB
    __bf16* xkv_bf = (__bf16*)(ws + 24 * MB);   // 16 MB
    __bf16* q_bf   = (__bf16*)(ws + 40 * MB);   // 16 MB  [n][q][d]
    __bf16* k_bf   = (__bf16*)(ws + 56 * MB);   // 16 MB  [n][k][d]
    __bf16* vT_bf  = (__bf16*)(ws + 72 * MB);   // 16 MB  [n][d][k]
    __bf16* ctx_bf = (__bf16*)(ws + 88 * MB);   // 16 MB  [t][b][h]

    // Stage 0: fold r into activations, cast everything to bf16
    cast_scale_x<<<Mrows * Hdim / 256, 256, 0, stream>>>(inputs_q, r_q, xq_bf);
    cast_scale_x<<<Mrows * Hdim / 256, 256, 0, stream>>>(inputs_kv, r_kv, xkv_bf);
    cast_w<<<Hdim * Hdim / 256, 256, 0, stream>>>(w_q, wq_bf, Hdim * Hdim);
    cast_w<<<2 * Hdim * Hdim / 256, 256, 0, stream>>>(w_kv, wkv_bf, 2 * Hdim * Hdim);
    cast_w<<<Hdim * Hdim / 256, 256, 0, stream>>>(w_o, wo_bf, Hdim * Hdim);

    // Stage 1: Q projection (M=8192, N=1024, K=1024)
    gemm_wmma<0><<<dim3(Mrows / 128, Hdim / 64), 256, 0, stream>>>(
        xq_bf, wq_bf, s_q, b_q, (void*)q_bf, nullptr, Hdim, Hdim);

    // Stage 2: KV projection (M=8192, N=2048, K=1024)
    gemm_wmma<1><<<dim3(Mrows / 128, 2 * Hdim / 64), 256, 0, stream>>>(
        xkv_bf, wkv_bf, s_kv, b_kv, (void*)k_bf, (void*)vT_bf, 2 * Hdim, Hdim);

    // Stage 3: flash attention over 128 heads x 64 q-tiles
    attn_wmma<<<dim3(NTOT, Tq / 16 / 8), 256, 0, stream>>>(q_bf, k_bf, vT_bf, ctx_bf);

    // Stage 4: output projection (M=8192, N=1024, K=1024), f32 out
    gemm_wmma<2><<<dim3(Mrows / 128, Hdim / 64), 256, 0, stream>>>(
        ctx_bf, wo_bf, nullptr, b_o, d_out, nullptr, Hdim, Hdim);
}
